// ClassTransformerLayer_25048249270632
// MI455X (gfx1250) — matrix operand order repaired and checked by reading, not compiled, not run on hardware
//
#include <hip/hip_runtime.h>
#include <math.h>

// ---------------- CDNA5 WMMA types & helpers (gfx1250, wave32) ----------------
typedef _Float16     v16h  __attribute__((ext_vector_type(16)));
typedef float        v8f   __attribute__((ext_vector_type(8)));
typedef unsigned int u32x4 __attribute__((ext_vector_type(4)));
typedef int          i32x4 __attribute__((ext_vector_type(4)));
typedef int          i32x8 __attribute__((ext_vector_type(8)));

#ifndef __has_builtin
#define __has_builtin(x) 0
#endif
#if __has_builtin(__builtin_amdgcn_tensor_load_to_lds)
#define HAVE_TDM 1
#else
#define HAVE_TDM 0
#endif

__device__ __forceinline__ v8f wmma_f16(v16h a, v16h b, v8f c) {
  // (neg_a, A, neg_b, B, c_mod, C, reuse_a, reuse_b)
  return __builtin_amdgcn_wmma_f32_16x16x32_f16(false, a, false, b, (short)0, c,
                                                false, false);
}

// A fragment: 16x32 f16 tile, row-major source with leading dim `ld`.
// lanes 0-15 = M rows; lane-half adds +16 to K;
// VGPR i holds K pairs {2i,2i+1} => element e is k = k0 + (laneHalf<<4) + e.
__device__ __forceinline__ v16h frag_a(const _Float16* base, int ld, int mt, int k0) {
  const int lane = threadIdx.x & 31;
  const _Float16* row = base + (size_t)(mt * 16 + (lane & 15)) * ld + k0 + ((lane >> 4) << 4);
  v16h a;
#pragma unroll
  for (int i = 0; i < 8; ++i) {
    int kb = i << 1;  // 0,2,4,...,14
    a[2 * i]     = row[kb];
    a[2 * i + 1] = row[kb + 1];
  }
  return a;
}

// B fragment from row-major [K][N] in LDS (used for vh / bd operands).
// lanes 0-15 = N cols, lane-half selects K+16; VGPR j holds rows 2j,2j+1
// => element e of the v16h corresponds to k = k0 + (laneHalf<<4) + e.
__device__ __forceinline__ v16h frag_b(const _Float16* base, int ld, int k0, int nt) {
  const int lane = threadIdx.x & 31;
  const _Float16* col = base + (size_t)(k0 + ((lane >> 4) << 4)) * ld + nt * 16 + (lane & 15);
  v16h b;
#pragma unroll
  for (int j = 0; j < 8; ++j) {
    b[2 * j]     = col[(size_t)(2 * j) * ld];
    b[2 * j + 1] = col[(size_t)(2 * j + 1) * ld];
  }
  return b;
}

// B fragment from fragment-ready (pre-swizzled) weights: one contiguous
// 32-byte per-lane load -> global_load_b128 x2, no half packing.
__device__ __forceinline__ v16h frag_bw(const _Float16* wfrag, int KT, int kt, int nt) {
  return *(const v16h*)(wfrag + (((nt * KT + kt) << 9) + ((threadIdx.x & 31) << 4)));
}

// ---------------- problem constants ----------------
#define BB 2
#define CC 64
#define TT 128      // real T
#define TP 256      // padded length (pad part handled analytically)
#define HH 64
#define WW 64
#define HP 16
#define WP 16
#define NSEQ (BB * HP * WP)   // 512

// workspace byte offsets (fragment-ready f16 weights, sizes unchanged)
#define WS_WQH 0u              // 16 tiles * 512 halves
#define WS_WKH 16384u
#define WS_WVH 32768u
#define WS_W1H 40960u
#define WS_W2H 73728u
#define WS_PKS 106496u         // f32 [64]
#define WS_PKV 106752u         // f32 [8][8][8]
#define WS_XS0 131072u         // f32 [512][128][64]

// dynamic LDS layout for the fused attention+MLP kernel
#define OFF_X    0u            // f32 [128][64]
#define OFF_A    32768u        // f16 [128][128] qk_in; stage2: f16 [128][64] xn2
#define OFF_HMID 49152u        // f16 [128][256] (stage2 only; overlays QH/KH/VH)
#define OFF_QH   65536u        // f16 [128][64]
#define OFF_KH   81920u        // f16 [128][64]
#define OFF_VH   98304u        // f16 [128][64]
#define OFF_BD   114688u       // f16 [64][64] block-diag KV
#define OFF_KS   122880u       // f32 [64]
#define OFF_Z    123136u       // f32 [128][8]
#define DYN_LDS  127232u

// ---------------- K0: fp32 -> fragment-ready f16 weight staging ----------------
// dst layout per GEMM weight: tile (nt*KT + kt), within tile [lane][e] with
//   k = kt*32 + (lane>>4)*16 + e ,  n = nt*16 + (lane&15)
__global__ void convert_weights_kernel(const float* __restrict__ Wq,
                                       const float* __restrict__ Wk,
                                       const float* __restrict__ Wv,
                                       const float* __restrict__ W1,
                                       const float* __restrict__ W2,
                                       _Float16* __restrict__ out) {
  int i = blockIdx.x * 256 + threadIdx.x;
  if (i >= 53248) return;
  const float* src; int K, N, li, base;
  if (i < 8192)       { src = Wq; K = 128; N = 64;  li = i;         base = 0;     }
  else if (i < 16384) { src = Wk; K = 128; N = 64;  li = i - 8192;  base = 8192;  }
  else if (i < 20480) { src = Wv; K = 64;  N = 64;  li = i - 16384; base = 16384; }
  else if (i < 36864) { src = W1; K = 64;  N = 256; li = i - 20480; base = 20480; }
  else                { src = W2; K = 256; N = 64;  li = i - 36864; base = 36864; }
  int KT = K >> 5;
  int tile = li >> 9, lane = (li >> 4) & 31, e = li & 15;
  int kt = tile % KT, nt = tile / KT;
  int k = (kt << 5) + ((lane >> 4) << 4) + e;
  int n = (nt << 4) + (lane & 15);
  out[base + li] = (_Float16)src[k * N + n];
}

// ---------------- K1: constant pad-token attention statistics ----------------
// KV_pad = (Tp-T) * k_pad (x) v_pad ; Ksum_pad = (Tp-T) * k_pad
__global__ void pad_const_kernel(const float* __restrict__ pad_tok,
                                 const float* __restrict__ pad_guid,
                                 const float* __restrict__ Wk, const float* __restrict__ bk,
                                 const float* __restrict__ Wv, const float* __restrict__ bv,
                                 const float* __restrict__ ln1g, const float* __restrict__ ln1b,
                                 float* __restrict__ padks, float* __restrict__ padkv) {
  __shared__ float pt[64], qkin[128], kp[64], vp[64];
  int t = threadIdx.x;            // 64 threads
  pt[t] = pad_tok[t];
  __syncthreads();
  float mu = 0.f;
  for (int i = 0; i < 64; ++i) mu += pt[i];
  mu *= (1.0f / 64.0f);
  float var = 0.f;
  for (int i = 0; i < 64; ++i) { float d = pt[i] - mu; var += d * d; }
  var *= (1.0f / 64.0f);
  float rstd = rsqrtf(var + 1e-5f);
  qkin[t]      = (pt[t] - mu) * rstd * ln1g[t] + ln1b[t];
  qkin[64 + t] = pad_guid[t];
  __syncthreads();
  float ka = bk[t];
  for (int i = 0; i < 128; ++i) ka += qkin[i] * Wk[i * 64 + t];
  ka = (ka > 0.f) ? (ka + 1.f) : __expf(ka);     // elu + 1
  float va = bv[t];
  for (int i = 0; i < 64; ++i) va += qkin[i] * Wv[i * 64 + t];
  va *= (1.0f / (float)TP);
  kp[t] = ka; vp[t] = va;
  padks[t] = (float)(TP - TT) * ka;
  __syncthreads();
  for (int p = t; p < 512; p += 64) {
    int h = p >> 6, d = (p >> 3) & 7, dv = p & 7;
    padkv[p] = (float)(TP - TT) * kp[h * 8 + d] * vp[h * 8 + dv];
  }
}

// ---------------- K2: 4x4 avg-pool + layout change ----------------
// x[B][C][T][H][W] -> xs0[n=(b,hp,wp)][t][c]
__global__ __launch_bounds__(256) void pool_kernel(const float* __restrict__ x,
                                                   float* __restrict__ xs0) {
  int lin = blockIdx.x * 256 + threadIdx.x;     // 4,194,304 total
  int wp = lin & 15, hp = (lin >> 4) & 15, t = (lin >> 8) & 127,
      c = (lin >> 15) & 63, b = lin >> 21;
  const float* base = x + ((((size_t)b * CC + c) * TT + t) * HH + hp * 4) * WW + wp * 4;
  __builtin_prefetch(base + 4 * WW, 0, 1);      // -> global_prefetch_b8
  const float4* src = (const float4*)base;
  float4 s0 = src[0], s1 = src[16], s2 = src[32], s3 = src[48];  // 4 rows
  float s = s0.x + s0.y + s0.z + s0.w + s1.x + s1.y + s1.z + s1.w +
            s2.x + s2.y + s2.z + s2.w + s3.x + s3.y + s3.z + s3.w;
  xs0[(((size_t)(b * HP + hp) * WP + wp) * TT + t) * CC + c] = s * 0.0625f;
}

// ---------------- K3: fused LN1 + linear attention + LN2 + MLP ----------------
// One block per sequence n (512 blocks, 256 threads = 8 waves). All 5 GEMMs via
// v_wmma_f32_16x16x32_f16; input tile staged by the Tensor Data Mover.
__global__ __launch_bounds__(256) void attn_mlp_kernel(
    float* __restrict__ xs0, const float* __restrict__ guidance,
    const _Float16* __restrict__ wqh, const _Float16* __restrict__ wkh,
    const _Float16* __restrict__ wvh, const _Float16* __restrict__ w1h,
    const _Float16* __restrict__ w2h,
    const float* __restrict__ bq, const float* __restrict__ bk,
    const float* __restrict__ bv, const float* __restrict__ b1,
    const float* __restrict__ b2,
    const float* __restrict__ ln1g, const float* __restrict__ ln1b,
    const float* __restrict__ ln2g, const float* __restrict__ ln2b,
    const float* __restrict__ padks, const float* __restrict__ padkv) {
  extern __shared__ char smem[];
  float*    sx  = (float*)(smem + OFF_X);      // [128][64] residual stream
  _Float16* sa  = (_Float16*)(smem + OFF_A);   // [128][128] qk_in
  _Float16* qh  = (_Float16*)(smem + OFF_QH);  // [128][64]
  _Float16* kh  = (_Float16*)(smem + OFF_KH);  // [128][64]
  _Float16* vh  = (_Float16*)(smem + OFF_VH);  // [128][64]
  _Float16* bd  = (_Float16*)(smem + OFF_BD);  // [64][64] block-diag KV
  float*    ks  = (float*)(smem + OFF_KS);     // [64]
  float*    zz  = (float*)(smem + OFF_Z);      // [128][8]
  _Float16* hm  = (_Float16*)(smem + OFF_HMID);// [128][256] MLP hidden

  const int tid  = threadIdx.x;
  const int lane = tid & 31;
  const int w    = tid >> 5;          // wave id 0..7
  const int n    = blockIdx.x;        // sequence
  const int b    = n >> 8;            // batch

  // ---- stage 128x64 f32 tile into LDS via TDM (tensor_load_to_lds) ----
#if HAVE_TDM
  if (w == 0) {
    unsigned long long ga = (unsigned long long)(xs0 + (size_t)n * TT * CC);
    u32x4 g0;
    g0[0] = 1u;                                        // count=1 (valid user D#)
    g0[1] = (unsigned)OFF_X;                           // lds_addr (bytes)
    g0[2] = (unsigned)(ga & 0xffffffffu);              // global_addr lo
    g0[3] = (unsigned)((ga >> 32) & 0x1ffffffu) | (2u << 30);  // hi + type=2
    i32x8 g1;
    g1[0] = (int)(2u << 16);     // data_size = 4B, no multicast/iterate/pad
    g1[1] = (int)(64u << 16);    // tensor_dim0 = 64       (bits 79:48 lo)
    g1[2] = (int)(128u << 16);   // tensor_dim1 = 128      (bits 111:80 lo)
    g1[3] = (int)(64u << 16);    // tile_dim0   = 64       (bits 127:112)
    g1[4] = 128;                 // tile_dim1   = 128      (bits 143:128)
    g1[5] = 64;                  // tensor_dim0_stride = 64 (bits 191:160)
    g1[6] = 0;
    g1[7] = 0;
    i32x4 z4 = {0, 0, 0, 0};
#if __clang_major__ >= 23
    i32x8 z8 = {0, 0, 0, 0, 0, 0, 0, 0};
    __builtin_amdgcn_tensor_load_to_lds(g0, g1, z4, z4, z8, 0);
#else
    __builtin_amdgcn_tensor_load_to_lds(g0, g1, z4, z4, 0);
#endif
    __builtin_amdgcn_s_wait_tensorcnt(0);
  }
#else
  {
    const float4* src = (const float4*)(xs0 + (size_t)n * TT * CC);
    float4* dst = (float4*)sx;
    for (int i = tid; i < (TT * CC) / 4; i += 256) dst[i] = src[i];
  }
#endif
  __syncthreads();

  // ---- LN1 -> sa[:, :64], guidance -> sa[:, 64:128] (f16) ----
  for (int r = w; r < TT; r += 8) {
    float v0 = sx[r * CC + lane], v1 = sx[r * CC + lane + 32];
    float s = v0 + v1;
#pragma unroll
    for (int o = 16; o; o >>= 1) s += __shfl_xor(s, o);
    float mu = s * (1.0f / 64.0f);
    float d0 = v0 - mu, d1 = v1 - mu;
    float q2 = d0 * d0 + d1 * d1;
#pragma unroll
    for (int o = 16; o; o >>= 1) q2 += __shfl_xor(q2, o);
    float rstd = rsqrtf(q2 * (1.0f / 64.0f) + 1e-5f);
    sa[r * 128 + lane]      = (_Float16)(d0 * rstd * ln1g[lane] + ln1b[lane]);
    sa[r * 128 + lane + 32] = (_Float16)(d1 * rstd * ln1g[lane + 32] + ln1b[lane + 32]);
    sa[r * 128 + 64 + lane] = (_Float16)guidance[((size_t)b * TT + r) * CC + lane];
    sa[r * 128 + 96 + lane] = (_Float16)guidance[((size_t)b * TT + r) * CC + lane + 32];
  }
  __syncthreads();

  // ---- Q = elu(qk_in @ Wq + bq)+1 ; K likewise ; vals = (xn @ Wv + bv)/Tp ----
  for (int nt = 0; nt < 4; ++nt) {          // Q
    v8f acc = {};
#pragma unroll
    for (int k0 = 0; k0 < 128; k0 += 32)
      acc = wmma_f16(frag_a(sa, 128, w, k0), frag_bw(wqh, 4, k0 >> 5, nt), acc);
#pragma unroll
    for (int i = 0; i < 8; ++i) {
      int r = w * 16 + ((lane >> 4) << 3) + i, c = nt * 16 + (lane & 15);
      float v = acc[i] + bq[c];
      qh[r * CC + c] = (_Float16)((v > 0.f) ? (v + 1.f) : __expf(v));
    }
  }
  for (int nt = 0; nt < 4; ++nt) {          // K
    v8f acc = {};
#pragma unroll
    for (int k0 = 0; k0 < 128; k0 += 32)
      acc = wmma_f16(frag_a(sa, 128, w, k0), frag_bw(wkh, 4, k0 >> 5, nt), acc);
#pragma unroll
    for (int i = 0; i < 8; ++i) {
      int r = w * 16 + ((lane >> 4) << 3) + i, c = nt * 16 + (lane & 15);
      float v = acc[i] + bk[c];
      kh[r * CC + c] = (_Float16)((v > 0.f) ? (v + 1.f) : __expf(v));
    }
  }
  for (int nt = 0; nt < 4; ++nt) {          // vals
    v8f acc = {};
#pragma unroll
    for (int k0 = 0; k0 < 64; k0 += 32)
      acc = wmma_f16(frag_a(sa, 128, w, k0), frag_bw(wvh, 2, k0 >> 5, nt), acc);
#pragma unroll
    for (int i = 0; i < 8; ++i) {
      int r = w * 16 + ((lane >> 4) << 3) + i, c = nt * 16 + (lane & 15);
      vh[r * CC + c] = (_Float16)((acc[i] + bv[c]) * (1.0f / (float)TP));
    }
  }
  // zero BD (off-diag tiles stay zero)
  for (int i = tid; i < 64 * 64; i += 256) bd[i] = (_Float16)0.f;
  __syncthreads();

  // ---- KV = K^T @ vals (diag 16x16 tiles only, waves 0-3) + Ksum (waves 4-5) ----
  if (w < 4) {
    int mt = w;
    v8f acc = {};
#pragma unroll
    for (int k0 = 0; k0 < 128; k0 += 32) {
      v16h a;  // transposed read: A[m][k] = K[k][m]
      int m = mt * 16 + (lane & 15), hs = (lane >> 4) << 4;
#pragma unroll
      for (int i = 0; i < 8; ++i) {
        int kb = k0 + (i << 1) + hs;
        a[2 * i]     = kh[kb * CC + m];
        a[2 * i + 1] = kh[(kb + 1) * CC + m];
      }
      acc = wmma_f16(a, frag_b(vh, 64, k0, mt), acc);
    }
#pragma unroll
    for (int i = 0; i < 8; ++i) {
      int r = mt * 16 + ((lane >> 4) << 3) + i, c = mt * 16 + (lane & 15);
      float v = 0.f;
      if ((r >> 3) == (c >> 3)) v = acc[i] + padkv[(r >> 3) * 64 + (r & 7) * 8 + (c & 7)];
      bd[r * 64 + c] = (_Float16)v;
    }
  } else if (w < 6) {
    int c = tid & 63;
    float s = padks[c];
    for (int l = 0; l < TT; ++l) s += (float)kh[l * CC + c];
    ks[c] = s;
  }
  __syncthreads();

  // ---- Z[l][h] = 1 / (Q_row_h . Ksum_h + eps) ----
  for (int p = tid; p < TT * 8; p += 256) {
    int l = p >> 3, h = p & 7;
    float s = 0.f;
#pragma unroll
    for (int d = 0; d < 8; ++d) s += (float)qh[l * CC + h * 8 + d] * ks[h * 8 + d];
    zz[p] = 1.0f / (s + 1e-6f);
  }
  __syncthreads();

  // ---- attn = (Q @ BD) * Tp * Z, residual into sx ----
  for (int nt = 0; nt < 4; ++nt) {
    v8f acc = {};
#pragma unroll
    for (int k0 = 0; k0 < 64; k0 += 32)
      acc = wmma_f16(frag_a(qh, 64, w, k0), frag_b(bd, 64, k0, nt), acc);
#pragma unroll
    for (int i = 0; i < 8; ++i) {
      int r = w * 16 + ((lane >> 4) << 3) + i, c = nt * 16 + (lane & 15);
      sx[r * CC + c] += acc[i] * (float)TP * zz[r * 8 + (c >> 3)];
    }
  }
  __syncthreads();

  // ---- LN2 -> sa (f16, ld 64) ----
  _Float16* sa2 = (_Float16*)(smem + OFF_A);
  for (int r = w; r < TT; r += 8) {
    float v0 = sx[r * CC + lane], v1 = sx[r * CC + lane + 32];
    float s = v0 + v1;
#pragma unroll
    for (int o = 16; o; o >>= 1) s += __shfl_xor(s, o);
    float mu = s * (1.0f / 64.0f);
    float d0 = v0 - mu, d1 = v1 - mu;
    float q2 = d0 * d0 + d1 * d1;
#pragma unroll
    for (int o = 16; o; o >>= 1) q2 += __shfl_xor(q2, o);
    float rstd = rsqrtf(q2 * (1.0f / 64.0f) + 1e-5f);
    sa2[r * 64 + lane]      = (_Float16)(d0 * rstd * ln2g[lane] + ln2b[lane]);
    sa2[r * 64 + lane + 32] = (_Float16)(d1 * rstd * ln2g[lane + 32] + ln2b[lane + 32]);
  }
  __syncthreads();

  // ---- MLP1: relu(xn2 @ W1 + b1) -> hm [128][256] ----
  for (int nt = 0; nt < 16; ++nt) {
    v8f acc = {};
#pragma unroll
    for (int k0 = 0; k0 < 64; k0 += 32)
      acc = wmma_f16(frag_a(sa2, 64, w, k0), frag_bw(w1h, 2, k0 >> 5, nt), acc);
#pragma unroll
    for (int i = 0; i < 8; ++i) {
      int r = w * 16 + ((lane >> 4) << 3) + i, c = nt * 16 + (lane & 15);
      float v = acc[i] + b1[c];
      hm[r * 256 + c] = (_Float16)(v > 0.f ? v : 0.f);
    }
  }
  __syncthreads();

  // ---- MLP2: hm @ W2 + b2, residual into sx ----
  for (int nt = 0; nt < 4; ++nt) {
    v8f acc = {};
#pragma unroll
    for (int k0 = 0; k0 < 256; k0 += 32)
      acc = wmma_f16(frag_a(hm, 256, w, k0), frag_bw(w2h, 8, k0 >> 5, nt), acc);
#pragma unroll
    for (int i = 0; i < 8; ++i) {
      int r = w * 16 + ((lane >> 4) << 3) + i, c = nt * 16 + (lane & 15);
      sx[r * CC + c] += acc[i] + b2[c];
    }
  }
  __syncthreads();

  // ---- write back (in place) ----
  {
    float4* dst = (float4*)(xs0 + (size_t)n * TT * CC);
    const float4* src = (const float4*)sx;
    for (int i = tid; i < (TT * CC) / 4; i += 256) dst[i] = src[i];
  }
}

// ---------------- K4: align-corners bilinear 16->64 upsample + residual ----------------
__global__ __launch_bounds__(256) void upsample_kernel(const float* __restrict__ x,
                                                       const float* __restrict__ xs,
                                                       float* __restrict__ out) {
  __shared__ float tile[256];
  int bid = blockIdx.x;                  // (b,t,c)
  int c = bid & 63, t = (bid >> 6) & 127, b = bid >> 13;
  int tid = threadIdx.x;
  int hp = tid >> 4, wp = tid & 15;
  tile[tid] = xs[(((size_t)(b * HP + hp) * WP + wp) * TT + t) * CC + c];
  __syncthreads();
  int tx = tid & 63, ty = tid >> 6;
  const float sc = 15.0f / 63.0f;
  float fx = tx * sc;
  int x0 = (int)fx; int x1 = (x0 + 1 < 15) ? x0 + 1 : 15; float wx = fx - (float)x0;
  size_t obase = (((size_t)b * CC + c) * TT + t) * (size_t)(HH * WW);
#pragma unroll
  for (int r = 0; r < 16; ++r) {
    int oy = ty * 16 + r;
    float fy = oy * sc;
    int y0 = (int)fy; int y1 = (y0 + 1 < 15) ? y0 + 1 : 15; float wy = fy - (float)y0;
    float v00 = tile[y0 * 16 + x0], v01 = tile[y0 * 16 + x1];
    float v10 = tile[y1 * 16 + x0], v11 = tile[y1 * 16 + x1];
    float val = (v00 * (1.f - wx) + v01 * wx) * (1.f - wy) +
                (v10 * (1.f - wx) + v11 * wx) * wy;
    size_t idx = obase + (size_t)oy * WW + tx;
    out[idx] = x[idx] + val;
  }
}

// ---------------- launcher ----------------
extern "C" void kernel_launch(void* const* d_in, const int* in_sizes, int n_in,
                              void* d_out, int out_size, void* d_ws, size_t ws_size,
                              hipStream_t stream) {
  (void)in_sizes; (void)n_in; (void)out_size; (void)ws_size;
  const float* x        = (const float*)d_in[0];
  const float* guidance = (const float*)d_in[1];
  const float* Wq  = (const float*)d_in[2];
  const float* bq  = (const float*)d_in[3];
  const float* Wk  = (const float*)d_in[4];
  const float* bk  = (const float*)d_in[5];
  const float* Wv  = (const float*)d_in[6];
  const float* bv  = (const float*)d_in[7];
  const float* W1  = (const float*)d_in[8];
  const float* b1  = (const float*)d_in[9];
  const float* W2  = (const float*)d_in[10];
  const float* b2  = (const float*)d_in[11];
  const float* ln1g = (const float*)d_in[12];
  const float* ln1b = (const float*)d_in[13];
  const float* ln2g = (const float*)d_in[14];
  const float* ln2b = (const float*)d_in[15];
  const float* pad_tok  = (const float*)d_in[16];
  const float* pad_guid = (const float*)d_in[17];

  char* ws = (char*)d_ws;
  _Float16* wh   = (_Float16*)(ws + WS_WQH);
  float* padks   = (float*)(ws + WS_PKS);
  float* padkv   = (float*)(ws + WS_PKV);
  float* xs0     = (float*)(ws + WS_XS0);
  float* out     = (float*)d_out;

  (void)hipFuncSetAttribute((const void*)attn_mlp_kernel,
                            hipFuncAttributeMaxDynamicSharedMemorySize, DYN_LDS);

  convert_weights_kernel<<<208, 256, 0, stream>>>(Wq, Wk, Wv, W1, W2, wh);
  pad_const_kernel<<<1, 64, 0, stream>>>(pad_tok, pad_guid, Wk, bk, Wv, bv,
                                         ln1g, ln1b, padks, padkv);
  pool_kernel<<<(BB * CC * TT * HP * WP) / 256, 256, 0, stream>>>(x, xs0);
  attn_mlp_kernel<<<NSEQ, 256, DYN_LDS, stream>>>(
      xs0, guidance,
      wh,                    // Wq tiles
      wh + 8192,             // Wk tiles
      wh + 16384,            // Wv tiles
      wh + 20480,            // W1 tiles
      wh + 36864,            // W2 tiles
      bq, bk, bv, b1, b2, ln1g, ln1b, ln2g, ln2b, padks, padkv);
  upsample_kernel<<<BB * TT * CC, 256, 0, stream>>>(x, xs0, out);
}
